// GraphConvolution_68247030334374
// MI455X (gfx1250) — matrix operand order repaired and checked by reading, not compiled, not run on hardware
//
#include <hip/hip_runtime.h>
#include <hip/hip_bf16.h>

// ---------------------------------------------------------------------------
// Relational graph conv for MI455X (gfx1250, wave32, WMMA + TDM).
//
// out[b,n,:] = relu( sum_dir sum_m vals_dir[b, m, rel(adj_dir[b,n,m]), :] )
// vals_dir[b,m,r,:] = (H+b_r) * sigmoid(G+bg_r), H=X W^T, G=H Wg^T
//
// Heavy step tensorized as one-hot GEMMs (v_wmma_f32_16x16x32_bf16) with
// 32x64 register tiles per wave (8 acc chains) and split-K(=8) over m.
// vals B-slabs are staged into LDS once per block by the Tensor Data Mover
// (TENSOR_LOAD_TO_LDS + s_wait_tensorcnt): the block's 4 waves share
// (b, dc-super, k-chunk) and differ in n-super, so one DMA feeds 4 waves.
// ---------------------------------------------------------------------------

typedef __attribute__((ext_vector_type(16))) __bf16 v16bf;
typedef __attribute__((ext_vector_type(8)))  __bf16 v8bf;
typedef __attribute__((ext_vector_type(8)))  float  v8f;
typedef __attribute__((ext_vector_type(4)))  unsigned int v4u;
typedef __attribute__((ext_vector_type(8)))  int v8i_;
typedef __attribute__((ext_vector_type(4)))  int v4i_;

#define BB    2
#define NN    512
#define DD    256
#define RELS  39
#define KSPL  8          // split-K chunks over m (each = 64 m's)
#define RC    13         // relations staged per TDM transfer (13*64 rows * 64B = 52KB)

// ---------------- f32 -> bf16 convert ----------------
__global__ void cvt_f32_bf16(const float* __restrict__ in,
                             __bf16* __restrict__ out, int n) {
    int i = blockIdx.x * blockDim.x + threadIdx.x;
    if (i < n) out[i] = (__bf16)in[i];
}

// ---------------- adj transpose via LDS tiles ----------------
__global__ void transpose_adj(const int* __restrict__ adj,
                              int* __restrict__ adjT) {
    __shared__ int tile[32][33];
    int b  = blockIdx.z;
    int x0 = blockIdx.x * 32, y0 = blockIdx.y * 32;
    const int* src = adj  + (size_t)b * NN * NN;
    int*       dst = adjT + (size_t)b * NN * NN;
#pragma unroll
    for (int ry = threadIdx.y; ry < 32; ry += 8)
        tile[ry][threadIdx.x] = src[(size_t)(y0 + ry) * NN + x0 + threadIdx.x];
    __syncthreads();
#pragma unroll
    for (int ry = threadIdx.y; ry < 32; ry += 8)
        dst[(size_t)(x0 + ry) * NN + y0 + threadIdx.x] = tile[threadIdx.x][ry];
}

// ---------------- C[M,256] = A_bf16[M,256] @ W^T (W row-major bf16) --------
__global__ void __launch_bounds__(128)
gemm_xwT(const __bf16* __restrict__ Xb, const __bf16* __restrict__ W,
         float* __restrict__ outF, __bf16* __restrict__ outB) {
    const int wave = threadIdx.x >> 5;
    const int lane = threadIdx.x & 31;
    const int half = lane >> 4;
    const int l16  = lane & 15;

    const int tile = blockIdx.x * 4 + wave;   // (M/16)*(D/16) tiles
    const int mt   = tile >> 4;               // D/16 == 16 column tiles
    const int nt   = tile & 15;

    const int m    = mt * 16 + l16;
    const int ncol = nt * 16 + l16;

    v8f acc = {};
#pragma unroll
    for (int k0 = 0; k0 < DD; k0 += 32) {
        const __bf16* xp = Xb + (size_t)m * DD + k0 + 16 * half;
        v8bf alo = *(const v8bf*)xp;
        v8bf ahi = *(const v8bf*)(xp + 8);
        v16bf a;
#pragma unroll
        for (int i = 0; i < 8; ++i) { a[i] = alo[i]; a[i + 8] = ahi[i]; }
        v16bf bfr = *(const v16bf*)(W + (size_t)ncol * DD + k0 + 16 * half);
        acc = __builtin_amdgcn_wmma_f32_16x16x32_bf16(
            false, a, false, bfr, (short)0, acc, false, false);
    }
    const int row0 = mt * 16 + 8 * half;
#pragma unroll
    for (int v = 0; v < 8; ++v) {
        float val = acc[v];
        outF[(size_t)(row0 + v) * DD + ncol] = val;
        if (outB) outB[(size_t)(row0 + v) * DD + ncol] = (__bf16)val;
    }
}

// ---------------- gated per-relation values, stored [b][r][dc][m] bf16 -----
__global__ void compute_vals(const float* __restrict__ H,
                             const float* __restrict__ G,
                             const float* __restrict__ bias,
                             const float* __restrict__ bg,
                             __bf16* __restrict__ vals, int R) {
    size_t idx   = (size_t)blockIdx.x * blockDim.x + threadIdx.x;
    size_t total = (size_t)BB * R * DD * NN;
    if (idx >= total) return;
    int m  = (int)(idx % NN);
    size_t t = idx / NN;
    int dc = (int)(t % DD);  t /= DD;
    int r  = (int)(t % R);
    int b  = (int)(t / R);
    float h = H[((size_t)b * NN + m) * DD + dc] + bias[(size_t)r * DD + dc];
    float g = G[((size_t)b * NN + m) * DD + dc] + bg[(size_t)r * DD + dc];
    float s = 1.0f / (1.0f + __expf(-g));
    vals[idx] = (__bf16)(h * s);
}

// ---- TDM: stage 2D slab vals[rbase..][dc0..+63][m0..+31] into LDS ---------
// rows = nrel_chunk*64 rows of 64B, row stride NN*2 bytes. data_size=4B units.
__device__ __forceinline__ void tdm_stage(const __bf16* vbase, int rbase,
                                          int rows, int dc0, int m0,
                                          unsigned ldsOff) {
    unsigned long long ga = (unsigned long long)(size_t)vbase
        + (((size_t)(rbase * DD + dc0) * NN) + m0) * 2;
    v4u g0;
    g0[0] = 1u;                                   // count=1, user descriptor
    g0[1] = ldsOff;                               // lds_addr (bytes)
    g0[2] = (unsigned)ga;                         // global_addr[31:0]
    g0[3] = (unsigned)(ga >> 32) | (2u << 30);    // global_addr[56:32] | type=2
    v8i_ g1;
    g1[0] = (2 << 16);                            // wg_mask=0, data_size=4B
    g1[1] = (16 << 16);                           // tensor_dim0 = 16 dwords
    g1[2] = (rows & 0xFFFF) << 16;                // tensor_dim1 lo
    g1[3] = ((rows >> 16) & 0xFFFF) | (16 << 16); // tensor_dim1 hi | tile_dim0=16
    g1[4] = rows & 0xFFFF;                        // tile_dim1=rows, tile_dim2=0
    g1[5] = 256;                                  // tensor_dim0_stride = 256 dwords
    g1[6] = 0;                                    // stride hi | dim1_stride lo
    g1[7] = 0;
    v4i_ z = { 0, 0, 0, 0 };
#if defined(__clang_major__) && __clang_major__ >= 23
    v8i_ z8 = { 0, 0, 0, 0, 0, 0, 0, 0 };
    __builtin_amdgcn_tensor_load_to_lds(g0, g1, z, z, z8, 0);
#else
    __builtin_amdgcn_tensor_load_to_lds(g0, g1, z, z, 0);
#endif
}

// ---- one direction, 32x64 register tile, TDM-staged B slabs ---------------
__device__ __forceinline__ void rel_dir_staged(
    const int* __restrict__ adjB,       // adj(+b offset), row-major [N][N]
    const __bf16* __restrict__ vbase,   // vals(+b offset), [r][dc][m]
    int nrel, int relBase, int n0, int dc0, int mstart,
    int l16, int half, int wave, __bf16* sbuf, unsigned ldsOff,
    v8f acc[2][4]) {
    const int* row0p = adjB + (size_t)(n0 + l16) * NN;
    const int* row1p = adjB + (size_t)(n0 + 16 + l16) * NN;
#pragma unroll 1
    for (int mi = 0; mi < 2; ++mi) {
        const int m0 = mstart + 32 * mi;
        if (mi == 0) {  // prefetch next adj chunk (global_prefetch_b8)
            __builtin_prefetch(row0p + m0 + 32 + 8 * half, 0, 3);
            __builtin_prefetch(row1p + m0 + 32 + 8 * half, 0, 3);
        }
        // per-lane A strips for both 16-row groups (held across r-chunks)
        int4 a0 = *(const int4*)(row0p + m0 + 16 * half);
        int4 a1 = *(const int4*)(row0p + m0 + 16 * half + 4);
        int4 a2 = *(const int4*)(row0p + m0 + 16 * half + 8);
        int4 a3 = *(const int4*)(row0p + m0 + 16 * half + 12);
        int4 b0 = *(const int4*)(row1p + m0 + 16 * half);
        int4 b1 = *(const int4*)(row1p + m0 + 16 * half + 4);
        int4 b2 = *(const int4*)(row1p + m0 + 16 * half + 8);
        int4 b3 = *(const int4*)(row1p + m0 + 16 * half + 12);
        int alo[8] = { a0.x,a0.y,a0.z,a0.w, a1.x,a1.y,a1.z,a1.w };
        int ahi[8] = { a2.x,a2.y,a2.z,a2.w, a3.x,a3.y,a3.z,a3.w };
        int blo[8] = { b0.x,b0.y,b0.z,b0.w, b1.x,b1.y,b1.z,b1.w };
        int bhi[8] = { b2.x,b2.y,b2.z,b2.w, b3.x,b3.y,b3.z,b3.w };
#pragma unroll 1
        for (int rb = 0; rb < nrel; rb += RC) {
            const int rend = (rb + RC < nrel) ? rb + RC : nrel;
            const int rows = (rend - rb) * 64;
            if (wave == 0) {                       // one DMA per block
                tdm_stage(vbase, rb, rows, dc0, m0, ldsOff);
                __builtin_amdgcn_s_wait_tensorcnt(0);
            }
            __syncthreads();                       // slab visible to all waves
#pragma unroll 1
            for (int r = rb; r < rend; ++r) {
                const int rv = relBase + r;
                v16bf af0, af1;
#pragma unroll
                for (int i = 0; i < 8; ++i) {
                    af0[i]     = (alo[i] == rv) ? (__bf16)1.0f : (__bf16)0.0f;
                    af0[i + 8] = (ahi[i] == rv) ? (__bf16)1.0f : (__bf16)0.0f;
                    af1[i]     = (blo[i] == rv) ? (__bf16)1.0f : (__bf16)0.0f;
                    af1[i + 8] = (bhi[i] == rv) ? (__bf16)1.0f : (__bf16)0.0f;
                }
                const __bf16* vr = sbuf + ((size_t)(r - rb) * 64 + l16) * 32
                                        + 16 * half;
#pragma unroll
                for (int j = 0; j < 4; ++j) {
                    // B fragment from LDS: row (rloc*64 + 16j + l16), 32B
                    v16bf bfr = *(const v16bf*)(vr + (size_t)(16 * j) * 32);
                    acc[0][j] = __builtin_amdgcn_wmma_f32_16x16x32_bf16(
                        false, af0, false, bfr, (short)0, acc[0][j], false, false);
                    acc[1][j] = __builtin_amdgcn_wmma_f32_16x16x32_bf16(
                        false, af1, false, bfr, (short)0, acc[1][j], false, false);
                }
            }
            __syncthreads();                       // done reading before restage
        }
    }
}

// ---------------- relation-expanded GEMM, split-K partials -----------------
__global__ void __launch_bounds__(128)
rel_gemm(const int* __restrict__ adj, const int* __restrict__ adjT,
         const __bf16* __restrict__ valsF, const __bf16* __restrict__ valsB,
         const __bf16* __restrict__ valsL, float* __restrict__ part) {
    extern __shared__ __bf16 sbuf[];               // RC*64 rows x 64B
    const int wave = threadIdx.x >> 5;
    const int lane = threadIdx.x & 31;
    const int half = lane >> 4;
    const int l16  = lane & 15;

    // 256 blocks: kc(8) x dcs(4) x nsg(4) x b(2); block's 4 waves share
    // (b,dcs,kc) -> identical B slabs; wave picks n-super.
    const int t   = blockIdx.x;
    const int kc  = t & 7;
    const int dcs = (t >> 3) & 3;
    const int nsg = (t >> 5) & 3;
    const int b   = t >> 7;
    const int ns  = nsg * 4 + wave;

    const int n0     = ns * 32;
    const int dc0    = dcs * 64;
    const int mstart = kc * 64;
    const unsigned ldsOff = (unsigned)(size_t)(void*)sbuf;

    const int* adjB  = adj  + (size_t)b * NN * NN;
    const int* adjTB = adjT + (size_t)b * NN * NN;
    const __bf16* vF = valsF + (size_t)b * RELS * DD * NN;
    const __bf16* vB = valsB + (size_t)b * RELS * DD * NN;
    const __bf16* vL = valsL + (size_t)b * 1 * DD * NN;

    v8f acc[2][4] = {};
    rel_dir_staged(adjB,  vF, RELS, 2, n0, dc0, mstart, l16, half, wave, sbuf, ldsOff, acc);
    rel_dir_staged(adjTB, vB, RELS, 2, n0, dc0, mstart, l16, half, wave, sbuf, ldsOff, acc);
    rel_dir_staged(adjTB, vL, 1,    1, n0, dc0, mstart, l16, half, wave, sbuf, ldsOff, acc);

    // store partials: part[kc][b][row][col]
    float* p = part + (size_t)kc * (BB * NN * DD) + (size_t)b * NN * DD;
#pragma unroll
    for (int i = 0; i < 2; ++i) {
        const int row0 = n0 + 16 * i + 8 * half;
#pragma unroll
        for (int j = 0; j < 4; ++j) {
            const int col = dc0 + 16 * j + l16;
#pragma unroll
            for (int v = 0; v < 8; ++v)
                p[(size_t)(row0 + v) * DD + col] = acc[i][j][v];
        }
    }
}

// ---------------- sum split-K partials + ReLU ------------------------------
__global__ void reduce_relu(const float* __restrict__ part,
                            float* __restrict__ out) {
    int i = blockIdx.x * blockDim.x + threadIdx.x;
    const int tot = BB * NN * DD;
    if (i >= tot) return;
    float s = 0.0f;
#pragma unroll
    for (int k = 0; k < KSPL; ++k) s += part[(size_t)k * tot + i];
    out[i] = fmaxf(s, 0.0f);
}

// ---------------------------------------------------------------------------
extern "C" void kernel_launch(void* const* d_in, const int* in_sizes, int n_in,
                              void* d_out, int out_size, void* d_ws, size_t ws_size,
                              hipStream_t stream) {
    const float* node   = (const float*)d_in[0];
    const int*   adj    = (const int*)d_in[2];
    const float* W[3]   = { (const float*)d_in[3], (const float*)d_in[4], (const float*)d_in[5] };
    const float* Wg[3]  = { (const float*)d_in[6], (const float*)d_in[7], (const float*)d_in[8] };
    const float* bia[3] = { (const float*)d_in[9], (const float*)d_in[10], (const float*)d_in[11] };
    const float* bg[3]  = { (const float*)d_in[12], (const float*)d_in[13], (const float*)d_in[14] };
    float* out = (float*)d_out;

    // ---- workspace carve-up (256B aligned) ----
    size_t off = 0;
    auto carve = [&](size_t bytes) {
        void* p = (char*)d_ws + off;
        off += (bytes + 255) & ~(size_t)255;
        return p;
    };
    const size_t MROWS = (size_t)BB * NN;            // 1024
    __bf16* Xbf = (__bf16*)carve(MROWS * DD * 2);
    __bf16* Wb[3], *Wgb[3];
    float *Hf[3], *Gf[3];
    __bf16* Hb[3];
    for (int d = 0; d < 3; ++d) Wb[d]  = (__bf16*)carve((size_t)DD * DD * 2);
    for (int d = 0; d < 3; ++d) Wgb[d] = (__bf16*)carve((size_t)DD * DD * 2);
    for (int d = 0; d < 3; ++d) Hf[d]  = (float*)carve(MROWS * DD * 4);
    for (int d = 0; d < 3; ++d) Hb[d]  = (__bf16*)carve(MROWS * DD * 2);
    for (int d = 0; d < 3; ++d) Gf[d]  = (float*)carve(MROWS * DD * 4);
    int* adjT = (int*)carve((size_t)BB * NN * NN * 4);
    __bf16* valsF = (__bf16*)carve((size_t)BB * RELS * DD * NN * 2);
    __bf16* valsB = (__bf16*)carve((size_t)BB * RELS * DD * NN * 2);
    __bf16* valsL = (__bf16*)carve((size_t)BB * 1 * DD * NN * 2);
    float*  part  = (float*)carve((size_t)KSPL * BB * NN * DD * 4);
    (void)ws_size; (void)n_in; (void)in_sizes; (void)out_size;

    // ---- 1. convert node + weights to bf16 ----
    {
        int n = (int)(MROWS * DD);
        cvt_f32_bf16<<<(n + 255) / 256, 256, 0, stream>>>(node, Xbf, n);
        int nw = DD * DD;
        for (int d = 0; d < 3; ++d) {
            cvt_f32_bf16<<<(nw + 255) / 256, 256, 0, stream>>>(W[d], Wb[d], nw);
            cvt_f32_bf16<<<(nw + 255) / 256, 256, 0, stream>>>(Wg[d], Wgb[d], nw);
        }
    }

    // ---- 2. adj transpose ----
    transpose_adj<<<dim3(16, 16, 2), dim3(32, 8), 0, stream>>>(adj, adjT);

    // ---- 3. H = X W^T (f32 + bf16), then G = Hb Wg^T (f32) ----
    {
        const int blocks = (int)((MROWS / 16) * (DD / 16) / 4);  // 256
        for (int d = 0; d < 3; ++d)
            gemm_xwT<<<blocks, 128, 0, stream>>>(Xbf, Wb[d], Hf[d], Hb[d]);
        for (int d = 0; d < 3; ++d)
            gemm_xwT<<<blocks, 128, 0, stream>>>(Hb[d], Wgb[d], Gf[d], (__bf16*)nullptr);
    }

    // ---- 4. gated relation values ----
    {
        size_t tot = (size_t)BB * RELS * DD * NN;
        int blocks = (int)((tot + 255) / 256);
        compute_vals<<<blocks, 256, 0, stream>>>(Hf[0], Gf[0], bia[0], bg[0], valsF, RELS);
        compute_vals<<<blocks, 256, 0, stream>>>(Hf[1], Gf[1], bia[1], bg[1], valsB, RELS);
        size_t totL = (size_t)BB * 1 * DD * NN;
        compute_vals<<<(int)((totL + 255) / 256), 256, 0, stream>>>(
            Hf[2], Gf[2], bia[2], bg[2], valsL, 1);
    }

    // ---- 5. relation-expanded WMMA GEMM (TDM-staged, split-K) + reduce ----
    const size_t ldsBytes = (size_t)RC * 64 * 64;    // 53,248 B
    rel_gemm<<<256, 128, ldsBytes, stream>>>(adj, adjT, valsF, valsB, valsL, part);
    reduce_relu<<<(BB * NN * DD + 255) / 256, 256, 0, stream>>>(part, out);
}
